// MultiHeadedAttention_33019708571650
// MI455X (gfx1250) — matrix operand order repaired and checked by reading, not compiled, not run on hardware
//
#include <hip/hip_runtime.h>
#include <hip/hip_bf16.h>
#include <cmath>

typedef __attribute__((ext_vector_type(16))) __bf16 v16bf;
typedef __attribute__((ext_vector_type(8)))  __bf16 v8bf;
typedef __attribute__((ext_vector_type(8)))  float  v8f;

namespace {

constexpr int Bn = 16, T = 512, DQ = 4096, DK = 2048, H = 8;
constexpr int dq = DQ / H;   // 512 (== T, required by cross_time)
constexpr int dk = DK / H;   // 256

__device__ inline void st_elem(float* p, float v)  { *p = v; }
__device__ inline void st_elem(__bf16* p, float v) { *p = (__bf16)v; }

// ---- WMMA bf16 16x16x32 fragment loaders (wave32 ISA layouts) ----
// A 16x32: lane = 16*half + m ; element i -> k = (i%8) + 16*(i/8) + 8*half
// => per lane: two contiguous 8-element (16B) runs at k0+8*half and k0+16+8*half.
__device__ inline v16bf load_a_row(const __bf16* __restrict__ A, int lda, int row, int k0, int lane) {
  const int half = lane >> 4, m = lane & 15;
  const __bf16* rp = A + (size_t)(row + m) * lda + (k0 + half * 8);
  v8bf lo = *(const v8bf*)(rp);
  v8bf hi = *(const v8bf*)(rp + 16);
  return __builtin_shufflevector(lo, hi, 0, 1, 2, 3, 4, 5, 6, 7,
                                 8, 9, 10, 11, 12, 13, 14, 15);
}

// A stored transposed in memory: A_T[k][m] with row stride lda (strided gather)
__device__ inline v16bf load_a_col(const __bf16* __restrict__ A, int lda, int row, int k0, int lane) {
  const int half = lane >> 4, m = lane & 15;
  const __bf16* cp = A + (row + m);
  const int ka = k0 + half * 8;
  const int kb = k0 + 16 + half * 8;
  v16bf a;
#pragma unroll
  for (int i = 0; i < 8; ++i) a[i]     = cp[(size_t)(ka + i) * lda];
#pragma unroll
  for (int i = 0; i < 8; ++i) a[8 + i] = cp[(size_t)(kb + i) * lda];
  return a;
}

// B 32x16 from W[n][k] row-major (B[k][n] = W[n][k]):
// lane = 16*half + n ; element i -> k = (i%8) + 16*(i/8) + 8*half  => two 16B runs.
__device__ inline v16bf load_b_row(const __bf16* __restrict__ W, int ldb, int k0, int col, int lane) {
  const int half = lane >> 4, n = lane & 15;
  const __bf16* rp = W + (size_t)(col + n) * ldb + (k0 + half * 8);
  v8bf lo = *(const v8bf*)(rp);
  v8bf hi = *(const v8bf*)(rp + 16);
  return __builtin_shufflevector(lo, hi, 0, 1, 2, 3, 4, 5, 6, 7,
                                 8, 9, 10, 11, 12, 13, 14, 15);
}

// C[M,N] = scale * A[M,K] x B^T (+ bias[N]); batched over z (b=z/Hd, h=z%Hd).
// Block = 8 waves (2x4); wave tile 64x64 (16 f32 accumulators); K step 32.
// Grid must tile M by 128 and N by 256 exactly. All operands bf16.
// CT: store C transposed, i.e. C[n*ldc + m] (contiguous along m per lane).
template <typename TC, bool AT, bool CT, bool BIAS>
__global__ __launch_bounds__(256) void gemm_wmma(
    const __bf16* __restrict__ A, int lda, long long sAb, long long sAh,
    const __bf16* __restrict__ Bp, int ldb, long long sBb, long long sBh,
    TC* __restrict__ C, int ldc, long long sCb, long long sCh,
    const float* __restrict__ bias, float scale, int K, int Hd) {
  const int z = blockIdx.z;
  const int zb = z / Hd, zh = z % Hd;
  A  += (size_t)zb * sAb + (size_t)zh * sAh;
  Bp += (size_t)zb * sBb + (size_t)zh * sBh;
  C  += (size_t)zb * sCb + (size_t)zh * sCh;

  const int tid = threadIdx.x;
  const int wave = tid >> 5, lane = tid & 31;
  const int rowBase = blockIdx.y * 128 + (wave >> 2) * 64;
  const int colBase = blockIdx.x * 256 + (wave & 3) * 64;

  v8f acc[4][4] = {};

  for (int k0 = 0; k0 < K; k0 += 32) {
    v16bf af[4], bfr[4];
#pragma unroll
    for (int im = 0; im < 4; ++im) {
      if constexpr (AT) af[im] = load_a_col(A, lda, rowBase + im * 16, k0, lane);
      else              af[im] = load_a_row(A, lda, rowBase + im * 16, k0, lane);
    }
#pragma unroll
    for (int in = 0; in < 4; ++in)
      bfr[in] = load_b_row(Bp, ldb, k0, colBase + in * 16, lane);
#pragma unroll
    for (int im = 0; im < 4; ++im)
#pragma unroll
      for (int in = 0; in < 4; ++in)
        acc[im][in] = __builtin_amdgcn_wmma_f32_16x16x32_bf16(
            false, af[im], false, bfr[in], (short)0, acc[im][in], false, false);
  }

  // D layout: col n = lane&15, row m = j + 8*(lane>>4) within each 16x16 tile.
  const int half = lane >> 4, n = lane & 15;
#pragma unroll
  for (int in = 0; in < 4; ++in) {
    const int c = colBase + in * 16 + n;
    const float bv = BIAS ? bias[c] : 0.0f;
#pragma unroll
    for (int im = 0; im < 4; ++im) {
      if constexpr (CT) {
        // contiguous 8 elements along m: emit one 16B store (TC must be __bf16)
        const int r0 = rowBase + im * 16 + 8 * half;
        v8bf ov;
#pragma unroll
        for (int j = 0; j < 8; ++j) ov[j] = (__bf16)(acc[im][in][j] * scale + bv);
        *(v8bf*)((__bf16*)C + (size_t)c * ldc + r0) = ov;
      } else {
#pragma unroll
        for (int j = 0; j < 8; ++j) {
          const int r = rowBase + im * 16 + j + 8 * half;
          st_elem(&C[(size_t)r * ldc + c], acc[im][in][j] * scale + bv);
        }
      }
    }
  }
}

// Elementwise f32 -> bf16 (n must be a multiple of 8)
__global__ __launch_bounds__(256) void cvt_bf16_kernel(const float* __restrict__ in,
                                                       __bf16* __restrict__ out, int n8) {
  const int i = blockIdx.x * blockDim.x + threadIdx.x;
  if (i >= n8) return;
  const float4 a = *(const float4*)(in + (size_t)i * 8);
  const float4 b = *(const float4*)(in + (size_t)i * 8 + 4);
  v8bf o;
  o[0] = (__bf16)a.x; o[1] = (__bf16)a.y; o[2] = (__bf16)a.z; o[3] = (__bf16)a.w;
  o[4] = (__bf16)b.x; o[5] = (__bf16)b.y; o[6] = (__bf16)b.z; o[7] = (__bf16)b.w;
  *(v8bf*)(out + (size_t)i * 8) = o;
}

// key [Bn][T][DK] f32 -> keyT [Bn][DK][T] bf16 (LDS 32x32 tile transpose)
__global__ __launch_bounds__(256) void transpose_cvt_kernel(const float* __restrict__ in,
                                                            __bf16* __restrict__ out) {
  __shared__ float tile[32][33];
  const int b = blockIdx.z;
  const int s0 = blockIdx.x * 32;   // over T
  const int c0 = blockIdx.y * 32;   // over DK
  const int tx = threadIdx.x & 31;
  const int ty = threadIdx.x >> 5;  // 8 rows per pass
  const float* ip = in + (size_t)b * T * DK;
  __bf16* op = out + (size_t)b * DK * T;
#pragma unroll
  for (int r = ty; r < 32; r += 8)
    tile[r][tx] = ip[(size_t)(s0 + r) * DK + (c0 + tx)];
  __syncthreads();
#pragma unroll
  for (int r = ty; r < 32; r += 8)
    op[(size_t)(c0 + r) * T + (s0 + tx)] = (__bf16)tile[tx][r];
}

// scores[z][t][e] (f32) -> attnT[z][t][e] (bf16), softmax over t per (z,e);
// mask[b][t]==0 uses finite FLT_MIN-style fill (matches jnp.finfo(f32).min semantics:
// fully-masked columns degrade to uniform 1/T, partial masking underflows to 0).
__global__ __launch_bounds__(256) void softmax_mask_kernel(
    const float* __restrict__ scores, const int* __restrict__ mask,
    __bf16* __restrict__ attnT) {
  const int z = blockIdx.x;      // 0..127  (b*H + h)
  const int b = z >> 3;
  const int e = threadIdx.x;     // 0..255
  constexpr float NEG = -3.4028234663852886e38f;
  const float* s = scores + (size_t)z * (T * dk) + e;
  const int* mk = mask + b * T;
  float mx = NEG;
  for (int t = 0; t < T; ++t) {
    float v = (mk[t] == 0) ? NEG : s[(size_t)t * dk];
    mx = fmaxf(mx, v);
  }
  float sum = 0.0f;
  for (int t = 0; t < T; ++t) {
    float v = (mk[t] == 0) ? NEG : s[(size_t)t * dk];
    sum += __expf(v - mx);
  }
  const float inv = 1.0f / sum;
  __bf16* o = attnT + (size_t)z * (T * dk) + e;
  for (int t = 0; t < T; ++t) {
    float v = (mk[t] == 0) ? NEG : s[(size_t)t * dk];
    o[(size_t)t * dk] = (__bf16)(__expf(v - mx) * inv);
  }
}

} // namespace

extern "C" void kernel_launch(void* const* d_in, const int* in_sizes, int n_in,
                              void* d_out, int out_size, void* d_ws, size_t ws_size,
                              hipStream_t stream) {
  (void)in_sizes; (void)n_in; (void)out_size; (void)ws_size;
  const float* query = (const float*)d_in[0];
  const float* key   = (const float*)d_in[1];
  const float* value = (const float*)d_in[2];
  const int*   mask  = (const int*)d_in[3];
  const float* Wq = (const float*)d_in[4];
  const float* bq = (const float*)d_in[5];
  const float* Wv = (const float*)d_in[6];
  const float* bv = (const float*)d_in[7];
  const float* Wo = (const float*)d_in[8];
  const float* bo = (const float*)d_in[9];
  float* out = (float*)d_out;

  char* ws = (char*)d_ws;
  __bf16* qproj  = (__bf16*)(ws);                  // [8192,4096]       64 MiB
  __bf16* vprojT = (__bf16*)(ws + (64ll  << 20));  // [16][2048][512]   32 MiB
  float*  scores = (float*) (ws + (96ll  << 20));  // [128][512][256]   64 MiB
  __bf16* attnT  = (__bf16*)(ws + (160ll << 20));  // [128][512][256]   32 MiB
  __bf16* xws    = (__bf16*)(ws + (192ll << 20));  // [16][256][2048]   16 MiB
  __bf16* queryB = (__bf16*)(ws + (208ll << 20));  // [8192,4096]       64 MiB
  __bf16* WqB    = (__bf16*)(ws + (272ll << 20));  // [4096,4096]       32 MiB
  __bf16* valueB = (__bf16*)(ws + (304ll << 20));  // [8192,2048]       32 MiB
  __bf16* WvB    = (__bf16*)(ws + (336ll << 20));  // [2048,2048]        8 MiB
  __bf16* WoB    = (__bf16*)(ws + (344ll << 20));  // [2048,2048]        8 MiB
  __bf16* keyT   = (__bf16*)(ws + (352ll << 20));  // [16][2048][512]   32 MiB

  const float inv_sqrt_d = 0.04419417382415922f;   // 1/sqrt(512)

  // 0) one-time bf16 conversions (bandwidth-bound, ~0.5 GB total)
  auto cvt = [&](const float* src, __bf16* dst, long long n) {
    const int n8 = (int)(n / 8);
    cvt_bf16_kernel<<<dim3((n8 + 255) / 256), 256, 0, stream>>>(src, dst, n8);
  };
  cvt(query, queryB, (long long)Bn * T * DQ);
  cvt(Wq,    WqB,    (long long)DQ * DQ);
  cvt(value, valueB, (long long)Bn * T * DK);
  cvt(Wv,    WvB,    (long long)DK * DK);
  cvt(Wo,    WoB,    (long long)DK * DK);
  transpose_cvt_kernel<<<dim3(T / 32, DK / 32, Bn), 256, 0, stream>>>(key, keyT);

  // 1) qproj = query @ Wq^T + bq   [8192 x 4096], K=4096
  gemm_wmma<__bf16, false, false, true>
      <<<dim3(DQ / 256, (Bn * T) / 128, 1), 256, 0, stream>>>(
          queryB, DQ, 0, 0, WqB, DQ, 0, 0, qproj, DQ, 0, 0, bq, 1.0f, DQ, 1);

  // 2) vprojT[b][c][t] = (value @ Wv^T + bv)^T, batched over b; M=512, N=2048, K=2048
  gemm_wmma<__bf16, false, true, true>
      <<<dim3(DK / 256, T / 128, Bn), 256, 0, stream>>>(
          valueB, DK, (long long)T * DK, 0,
          WvB, DK, 0, 0,
          vprojT, T, (long long)DK * T, 0,
          bv, 1.0f, DK, 1);

  // 3) scores[b,h] = (q[b,h] @ k[b,h]) / sqrt(512)   128 x ([512x512]@[512x256])
  //    A = qproj[b, t, h*512+s] (lda=4096); B^T rows = keyT[b][h*256+kk][s] (ldb=512)
  gemm_wmma<float, false, false, false>
      <<<dim3(dk / 256, T / 128, Bn * H), 256, 0, stream>>>(
          qproj, DQ, (long long)T * DQ, (long long)dq,
          keyT, T, (long long)DK * T, (long long)dk * T,
          scores, dk, (long long)H * T * dk, (long long)T * dk,
          nullptr, inv_sqrt_d, dq, H);

  // 4) masked softmax over t; output kept in [t][e] layout (coalesced writes)
  softmax_mask_kernel<<<dim3(Bn * H), 256, 0, stream>>>(scores, mask, attnT);

  // 5) x[b,h] = attn[b,h] @ v[b,h]   128 x ([256x512]@[512x256])
  //    A = attnT (transposed storage, lda=256); B^T rows = vprojT[b][h*256+u][t]
  //    C written as x[b, e, h*256+u]
  gemm_wmma<__bf16, true, false, false>
      <<<dim3(dk / 256, dk / 128, Bn * H), 256, 0, stream>>>(
          attnT, dk, (long long)H * T * dk, (long long)T * dk,
          vprojT, T, (long long)DK * T, (long long)dk * T,
          xws, DK, (long long)dk * DK, (long long)dk,
          nullptr, 1.0f, T, H);

  // 6) out = x @ Wo^T + bo   [4096 x 2048], K=2048
  gemm_wmma<float, false, false, true>
      <<<dim3(DK / 256, (Bn * dk) / 128, 1), 256, 0, stream>>>(
          xws, DK, 0, 0, WoB, DK, 0, 0, out, DK, 0, 0, bo, 1.0f, DK, 1);
}
